// QEffMolmo2Attention_34059090657658
// MI455X (gfx1250) — compile-verified, matrix operand order checked by reading, not run on hardware
//
#include <hip/hip_runtime.h>
#include <hip/hip_bf16.h>
#include <math.h>

// ---------------- problem constants ----------------
#define BB   2
#define SS   2048
#define DD   4096
#define HH   32
#define KVH_ 8
#define DH_  128
#define KVD_ (KVH_ * DH_)          // 1024
#define EE   (DD + 2 * KVD_)       // 6144
#define EPS_ 1e-6f
#define SCALING_ 0.08838834764831845f   // 128^-0.5

typedef __bf16 bf16;
typedef __attribute__((ext_vector_type(16))) __bf16 v16bf;
typedef __attribute__((ext_vector_type(8)))  float  v8f;
typedef __attribute__((ext_vector_type(4)))  unsigned int u32x4;
typedef __attribute__((ext_vector_type(8)))  int i32x8;
typedef __attribute__((ext_vector_type(4)))  int i32x4;

#if defined(__has_builtin)
#if __has_builtin(__builtin_amdgcn_tensor_load_to_lds) && \
    __has_builtin(__builtin_amdgcn_s_wait_tensorcnt)
#define USE_TDM 1
#endif
#endif

static __device__ __forceinline__ v8f wmma_bf16(v16bf a, v16bf b, v8f c) {
    return __builtin_amdgcn_wmma_f32_16x16x32_bf16(false, a, false, b, (short)0, c, false, false);
}

// ---------------- elementwise casts ----------------
__global__ void cast_f32_to_bf16_kernel(const float* __restrict__ in,
                                        bf16* __restrict__ out, size_t n) {
    size_t i = (size_t)blockIdx.x * blockDim.x + threadIdx.x;
    size_t stride = (size_t)gridDim.x * blockDim.x;
    for (; i < n; i += stride) out[i] = (bf16)in[i];
}

// in: [rows, cols] f32 row-major -> out: [cols, rows] bf16 row-major
__global__ void transpose_cast_kernel(const float* __restrict__ in,
                                      bf16* __restrict__ out,
                                      int rows, int cols) {
    size_t total = (size_t)rows * cols;
    size_t i = (size_t)blockIdx.x * blockDim.x + threadIdx.x;
    size_t stride = (size_t)gridDim.x * blockDim.x;
    for (; i < total; i += stride) {
        size_t d = i / cols;
        size_t e = i - d * cols;
        out[e * (size_t)rows + d] = (bf16)in[i];
    }
}

// ---------------- bf16 WMMA GEMM: C[M,N](f32) = A[M,K](bf16) * BT[N,K](bf16)^T
// block = 256 threads = 8 waves arranged 2(M) x 4(N);
// wave tile = 32(M) x 64(N) via 8 wmma accumulators; block tile = 64 x 256.
// Register double-buffering: fragments for k+32 are loaded while the 8 WMMAs
// for k execute, so s_wait_loadcnt covers a full iteration of latency.
__global__ __launch_bounds__(256) void gemm_bf16_kernel(
        const bf16* __restrict__ A, const bf16* __restrict__ BT,
        float* __restrict__ C, int M, int N, int K) {
    const int wave = threadIdx.x >> 5;
    const int lane = threadIdx.x & 31;
    const int lrow = lane & 15;
    const int lhi  = lane >> 4;
    const int wrow = wave >> 2;          // 0..1
    const int wcol = wave & 3;           // 0..3
    const int m0 = blockIdx.y * 64  + wrow * 32;
    const int n0 = blockIdx.x * 256 + wcol * 64;

    const bf16* ap0 = A  + (size_t)(m0 + lrow)      * K + lhi * 16;
    const bf16* ap1 = A  + (size_t)(m0 + 16 + lrow) * K + lhi * 16;
    const bf16* bp0 = BT + (size_t)(n0 +  0 + lrow) * K + lhi * 16;
    const bf16* bp1 = BT + (size_t)(n0 + 16 + lrow) * K + lhi * 16;
    const bf16* bp2 = BT + (size_t)(n0 + 32 + lrow) * K + lhi * 16;
    const bf16* bp3 = BT + (size_t)(n0 + 48 + lrow) * K + lhi * 16;

    v8f acc[2][4] = {};

    v16bf a0 = *(const v16bf*)(ap0);
    v16bf a1 = *(const v16bf*)(ap1);
    v16bf b0 = *(const v16bf*)(bp0);
    v16bf b1 = *(const v16bf*)(bp1);
    v16bf b2 = *(const v16bf*)(bp2);
    v16bf b3 = *(const v16bf*)(bp3);

    for (int k0 = 32; k0 < K; k0 += 32) {
        // issue next iteration's fragment loads first
        v16bf na0 = *(const v16bf*)(ap0 + k0);
        v16bf na1 = *(const v16bf*)(ap1 + k0);
        v16bf nb0 = *(const v16bf*)(bp0 + k0);
        v16bf nb1 = *(const v16bf*)(bp1 + k0);
        v16bf nb2 = *(const v16bf*)(bp2 + k0);
        v16bf nb3 = *(const v16bf*)(bp3 + k0);
        if (k0 + 512 < K) __builtin_prefetch(ap0 + k0 + 512, 0, 3);

        acc[0][0] = wmma_bf16(a0, b0, acc[0][0]);
        acc[0][1] = wmma_bf16(a0, b1, acc[0][1]);
        acc[0][2] = wmma_bf16(a0, b2, acc[0][2]);
        acc[0][3] = wmma_bf16(a0, b3, acc[0][3]);
        acc[1][0] = wmma_bf16(a1, b0, acc[1][0]);
        acc[1][1] = wmma_bf16(a1, b1, acc[1][1]);
        acc[1][2] = wmma_bf16(a1, b2, acc[1][2]);
        acc[1][3] = wmma_bf16(a1, b3, acc[1][3]);

        a0 = na0; a1 = na1;
        b0 = nb0; b1 = nb1; b2 = nb2; b3 = nb3;
    }
    // tail compute for the final fragments
    acc[0][0] = wmma_bf16(a0, b0, acc[0][0]);
    acc[0][1] = wmma_bf16(a0, b1, acc[0][1]);
    acc[0][2] = wmma_bf16(a0, b2, acc[0][2]);
    acc[0][3] = wmma_bf16(a0, b3, acc[0][3]);
    acc[1][0] = wmma_bf16(a1, b0, acc[1][0]);
    acc[1][1] = wmma_bf16(a1, b1, acc[1][1]);
    acc[1][2] = wmma_bf16(a1, b2, acc[1][2]);
    acc[1][3] = wmma_bf16(a1, b3, acc[1][3]);

#pragma unroll
    for (int wi = 0; wi < 2; ++wi) {
#pragma unroll
        for (int j = 0; j < 4; ++j) {
#pragma unroll
            for (int i = 0; i < 8; ++i) {
                int row = m0 + wi * 16 + i + lhi * 8;   // C layout: vgpr i -> M=i / i+8
                int col = n0 + j * 16 + lrow;
                C[(size_t)row * N + col] = acc[wi][j][i];
            }
        }
    }
}

// ---------------- RMSNorm + RoPE + V pack ----------------
__global__ __launch_bounds__(256) void rmsnorm_rope_kernel(
        const float* __restrict__ qkv,
        const float* __restrict__ q_scale, const float* __restrict__ k_scale,
        const float* __restrict__ cosc, const float* __restrict__ sinc,
        bf16* __restrict__ q_rot, bf16* __restrict__ k_rot, bf16* __restrict__ vT) {
    const int bs = blockIdx.x;
    const int b = bs / SS;
    const int s = bs - b * SS;
    const int tid = threadIdx.x;
    const float* row = qkv + (size_t)bs * EE;
    const float* cp = cosc + ((size_t)b * SS + s) * DH_;
    const float* sp = sinc + ((size_t)b * SS + s) * DH_;

    __shared__ float red[256];

    float acc = 0.f;
    for (int i = tid; i < DD; i += 256) { float x = row[i]; acc += x * x; }
    red[tid] = acc;
    __syncthreads();
    for (int o = 128; o > 0; o >>= 1) {
        if (tid < o) red[tid] += red[tid + o];
        __syncthreads();
    }
    const float qr = rsqrtf(red[0] / (float)DD + EPS_);
    __syncthreads();

    acc = 0.f;
    for (int i = tid; i < KVD_; i += 256) { float x = row[DD + i]; acc += x * x; }
    red[tid] = acc;
    __syncthreads();
    for (int o = 128; o > 0; o >>= 1) {
        if (tid < o) red[tid] += red[tid + o];
        __syncthreads();
    }
    const float kr = rsqrtf(red[0] / (float)KVD_ + EPS_);

    for (int i = tid; i < DD; i += 256) {
        int dh = i & (DH_ - 1);
        int h  = i >> 7;
        int pdh = (dh < 64) ? dh + 64 : dh - 64;
        int pi  = (i - dh) + pdh;
        float xn = row[i]  * qr * q_scale[i];
        float pn = row[pi] * qr * q_scale[pi];
        float rv = (dh < 64) ? -pn : pn;
        float o = xn * cp[dh] + rv * sp[dh];
        q_rot[(((size_t)b * HH + h) * SS + s) * DH_ + dh] = (bf16)o;
    }
    for (int i = tid; i < KVD_; i += 256) {
        int dh = i & (DH_ - 1);
        int kv = i >> 7;
        int pdh = (dh < 64) ? dh + 64 : dh - 64;
        int pi  = (i - dh) + pdh;
        float xn = row[DD + i]  * kr * k_scale[i];
        float pn = row[DD + pi] * kr * k_scale[pi];
        float rv = (dh < 64) ? -pn : pn;
        float o = xn * cp[dh] + rv * sp[dh];
        k_rot[(((size_t)b * KVH_ + kv) * SS + s) * DH_ + dh] = (bf16)o;
    }
    for (int i = tid; i < KVD_; i += 256) {
        int dh = i & (DH_ - 1);
        int kv = i >> 7;
        vT[(((size_t)b * KVH_ + kv) * DH_ + dh) * SS + s] = (bf16)row[DD + KVD_ + i];
    }
}

#if USE_TDM
// Issue a TDM load of a 32x128 bf16 tile (rows x DH) from global into LDS.
// Builds the D# descriptor per CDNA5 ISA ch.8: group0 = {ctrl, lds_addr,
// global_addr lo, global_addr hi | type=2}; group1 carries data_size=2B,
// tensor_dim0=128, tile 128x32, row stride 128.  All args are wave-uniform.
// This toolchain exposes the 6-argument builtin (extra i32x8 group + cpol).
static __device__ __forceinline__ void tdm_load_k_block(
        const bf16* gsrc, bf16* ldst, int rows_avail) {
    unsigned long long ga = (unsigned long long)(uintptr_t)gsrc;
    unsigned int la = (unsigned int)(uintptr_t)ldst;   // low 32 bits = LDS offset
    u32x4 g0;
    g0[0] = 1u;                                         // count=1, user D#
    g0[1] = la;                                         // lds_addr
    g0[2] = (unsigned int)ga;                           // global_addr[31:0]
    g0[3] = ((unsigned int)(ga >> 32) & 0x01FFFFFFu)    // global_addr[56:32]
            | (2u << 30);                               // type = 2 ("image")
    i32x8 g1;
    g1[0] = (1 << 16);                                  // data_size = 1 -> 2 bytes
    g1[1] = (DH_ & 0xFFFF) << 16;                       // tensor_dim0 lo16 = 128
    g1[2] = (rows_avail & 0xFFFF) << 16;                // dim0 hi16=0 | tensor_dim1 lo16
    g1[3] = ((rows_avail >> 16) & 0xFFFF)               // tensor_dim1 hi16
            | (DH_ << 16);                              // tile_dim0 = 128
    g1[4] = 32;                                         // tile_dim1 = 32 rows, tile_dim2=0
    g1[5] = DH_;                                        // tensor_dim0_stride lo32 = 128
    g1[6] = 0;
    g1[7] = 0;
    i32x4 z4 = {0, 0, 0, 0};
    i32x8 z8 = {0, 0, 0, 0, 0, 0, 0, 0};
    __builtin_amdgcn_tensor_load_to_lds(g0, g1, z4, z4, z8, 0);
}
#endif

// ---------------- flash attention (causal, GQA) ----------------
// block = 128 threads = 4 waves; each wave owns one 16-row q tile.
// K blocks staged to LDS with the Tensor Data Mover (double buffered,
// s_wait_tensorcnt) when available; QK^T and PV via v_wmma_f32_16x16x32_bf16;
// online softmax with ds_bpermute row broadcast.
__global__ __launch_bounds__(128) void flash_attn_kernel(
        const bf16* __restrict__ q_rot, const bf16* __restrict__ k_rot,
        const bf16* __restrict__ vT, bf16* __restrict__ attn_out) {
    const int wave = threadIdx.x >> 5;
    const int lane = threadIdx.x & 31;
    const int lrow = lane & 15;
    const int lhi  = lane >> 4;

    const int tiles_per_bh = SS / 64;
    const int bh = blockIdx.x / tiles_per_bh;
    const int qt = blockIdx.x - bh * tiles_per_bh;
    const int b = bh / HH;
    const int h = bh - b * HH;
    const int kvh = h >> 2;                  // H / KVH == 4
    const int qbase = qt * 64 + wave * 16;

    __shared__ __align__(16) float sS[4][16][33];
    __shared__ __align__(32) __bf16 sP[4][16][32];
#if USE_TDM
    __shared__ __align__(32) __bf16 kStage[4][2][32 * DH_];   // 8KB per buffer
#endif

    const bf16* qptr = q_rot + (((size_t)b * HH + h) * SS + qbase) * DH_;
    const bf16* kptr = k_rot + (((size_t)b * KVH_ + kvh) * SS) * DH_;
    const bf16* vptr = vT    + (((size_t)b * KVH_ + kvh) * DH_) * SS;

    v16bf qf[4];
#pragma unroll
    for (int kk = 0; kk < 4; ++kk)
        qf[kk] = *(const v16bf*)(qptr + (size_t)lrow * DH_ + kk * 32 + lhi * 16);

    v8f O[8] = {};
    float m_i = -1e30f;
    float l_i = 0.f;

    const int nkb = (qbase + 47) >> 5;       // causal: only key blocks <= row max

#if USE_TDM
    tdm_load_k_block(kptr, &kStage[wave][0][0], SS);
#endif

    for (int kb = 0; kb < nkb; ++kb) {
        const int tbase = kb * 32;

#if USE_TDM
        if (kb + 1 < nkb) {
            tdm_load_k_block(kptr + (size_t)(tbase + 32) * DH_,
                             &kStage[wave][(kb + 1) & 1][0], SS - tbase - 32);
            __builtin_amdgcn_s_wait_tensorcnt(1);
        } else {
            __builtin_amdgcn_s_wait_tensorcnt(0);
        }
        asm volatile("" ::: "memory");
        const __bf16* kbuf = &kStage[wave][kb & 1][0];
#endif

        // ---- scores: two 16x16 tiles covering 32 keys ----
#pragma unroll
        for (int tc = 0; tc < 2; ++tc) {
            v8f sc = {};
#pragma unroll
            for (int kk = 0; kk < 4; ++kk) {
#if USE_TDM
                v16bf kf = *(const v16bf*)(kbuf +
                    (size_t)(tc * 16 + lrow) * DH_ + kk * 32 + lhi * 16);
#else
                v16bf kf = *(const v16bf*)(kptr +
                    (size_t)(tbase + tc * 16 + lrow) * DH_ + kk * 32 + lhi * 16);
#endif
                sc = wmma_bf16(qf[kk], kf, sc);
            }
#pragma unroll
            for (int i = 0; i < 8; ++i)
                sS[wave][i + lhi * 8][tc * 16 + lrow] = sc[i];
        }
        __builtin_amdgcn_wave_barrier();

        // ---- row-wise online softmax (lane mirrors row lrow; uniform EXEC) ----
        const int qrow = qbase + lrow;
        float sv[32];
        float mnew = m_i;
#pragma unroll
        for (int c = 0; c < 32; ++c) {
            int t = tbase + c;
            float x = (t <= qrow) ? sS[wave][lrow][c] * SCALING_ : -10000.f;
            sv[c] = x;
            mnew = fmaxf(mnew, x);
        }
        const float alpha = __expf(m_i - mnew);
        float ls = 0.f;
#pragma unroll
        for (int c = 0; c < 32; ++c) {
            float p = __expf(sv[c] - mnew);
            ls += p;
            sP[wave][lrow][c] = (__bf16)p;
        }
        l_i = l_i * alpha + ls;
        m_i = mnew;
        __builtin_amdgcn_wave_barrier();

        // ---- rescale O by per-row alpha (broadcast from lane==row) ----
#pragma unroll
        for (int i = 0; i < 8; ++i) {
            int orow = i + lhi * 8;
            float a = __int_as_float(
                __builtin_amdgcn_ds_bpermute(orow << 2, __float_as_int(alpha)));
#pragma unroll
            for (int j = 0; j < 8; ++j) O[j][i] *= a;
        }

        // ---- PV: P(16x32) x V(32x16) per 16-wide dh tile ----
        v16bf pf = *(const v16bf*)(&sP[wave][lrow][lhi * 16]);
#pragma unroll
        for (int j = 0; j < 8; ++j) {
            v16bf vf = *(const v16bf*)(vptr +
                (size_t)(j * 16 + lrow) * SS + tbase + lhi * 16);
            O[j] = wmma_bf16(pf, vf, O[j]);
        }
    }

    // ---- epilogue: O / l, write bf16 [b, s, h*DH + dh] for the final GEMM ----
#pragma unroll
    for (int i = 0; i < 8; ++i) {
        int orow = i + lhi * 8;
        float l = __int_as_float(
            __builtin_amdgcn_ds_bpermute(orow << 2, __float_as_int(l_i)));
        float linv = 1.f / l;
#pragma unroll
        for (int j = 0; j < 8; ++j) {
            float val = O[j][i] * linv;
            attn_out[((size_t)b * SS + qbase + orow) * DD + h * DH_ + j * 16 + lrow] =
                (bf16)val;
        }
    }
}

// ---------------- host launcher ----------------
extern "C" void kernel_launch(void* const* d_in, const int* in_sizes, int n_in,
                              void* d_out, int out_size, void* d_ws, size_t ws_size,
                              hipStream_t stream) {
    const float* hs      = (const float*)d_in[0];   // [B,S,D]
    const float* w_qkv   = (const float*)d_in[1];   // [D,E]
    const float* w_out   = (const float*)d_in[2];   // [D,D]
    const float* q_scale = (const float*)d_in[3];   // [D]
    const float* k_scale = (const float*)d_in[4];   // [KVD]
    const float* cosc    = (const float*)d_in[5];   // [B,S,DH]
    const float* sinc    = (const float*)d_in[6];   // [B,S,DH]
    // d_in[7] attention_mask: causal, derived analytically in-kernel.
    float* out = (float*)d_out;                     // [B,S,D]

    char* ws = (char*)d_ws;
    size_t off = 0;
    bf16* hsb     = (bf16*)(ws + off); off += (size_t)BB * SS * DD * 2;        // 32 MB
    bf16* wqkvT   = (bf16*)(ws + off); off += (size_t)EE * DD * 2;             // 48 MB
    bf16* woutT   = (bf16*)(ws + off); off += (size_t)DD * DD * 2;             // 32 MB
    float* qkv    = (float*)(ws + off); off += (size_t)BB * SS * EE * 4;       // 96 MB
    bf16* q_rot   = (bf16*)(ws + off); off += (size_t)BB * HH * SS * DH_ * 2;  // 32 MB
    bf16* k_rot   = (bf16*)(ws + off); off += (size_t)BB * KVH_ * SS * DH_ * 2;// 8 MB
    bf16* vT      = (bf16*)(ws + off); off += (size_t)BB * KVH_ * SS * DH_ * 2;// 8 MB
    bf16* att_o   = (bf16*)(ws + off); off += (size_t)BB * SS * DD * 2;        // 32 MB

    const int M = BB * SS;   // 4096 rows

    cast_f32_to_bf16_kernel<<<2048, 256, 0, stream>>>(hs, hsb, (size_t)M * DD);
    transpose_cast_kernel<<<2048, 256, 0, stream>>>(w_qkv, wqkvT, DD, EE);
    transpose_cast_kernel<<<2048, 256, 0, stream>>>(w_out, woutT, DD, DD);
    gemm_bf16_kernel<<<dim3(EE / 256, M / 64), 256, 0, stream>>>(
        hsb, wqkvT, qkv, M, EE, DD);
    rmsnorm_rope_kernel<<<BB * SS, 256, 0, stream>>>(
        qkv, q_scale, k_scale, cosc, sinc, q_rot, k_rot, vT);
    flash_attn_kernel<<<BB * HH * (SS / 64), 128, 0, stream>>>(
        q_rot, k_rot, vT, att_o);
    gemm_bf16_kernel<<<dim3(DD / 256, M / 64), 256, 0, stream>>>(
        att_o, woutT, out, M, DD, DD);
}
